// MultiHeadLinearSelfAttention_62612033241692
// MI455X (gfx1250) — matrix operand order repaired and checked by reading, not compiled, not run on hardware
//
#include <hip/hip_runtime.h>
#include <hip/hip_bf16.h>
#include <stdint.h>

// ---------------------------------------------------------------------------
// Multi-head linear self-attention for MI455X (gfx1250, wave32, WMMA).
// All GEMMs via v_wmma_f32_16x16x32_bf16 (bf16 inputs, f32 accumulate).
// LDS tiles staged with GLOBAL_LOAD_ASYNC_TO_LDS_B128 (ASYNCcnt) + dbl buffer.
// ---------------------------------------------------------------------------

#define FEATS 768
#define HEADS 12
#define HDIM  64
#define BATCH 4
#define SEQ   4096
#define MROWS (BATCH * SEQ)      // 16384
#define BHC   (BATCH * HEADS)    // 48

typedef __bf16 bf16;
typedef __attribute__((ext_vector_type(16))) __bf16 v16bf;
typedef __attribute__((ext_vector_type(8)))  float  v8f;
typedef __attribute__((ext_vector_type(4)))  unsigned int u32x4;

union Frag16 {
    v16bf v;
    bf16  e[16];
    u32x4 q[2];
};

static __device__ __forceinline__ float bf2f(bf16 x) { return (float)x; }
static __device__ __forceinline__ bf16  f2bf(float x) { return (bf16)x; }

// phi(t) = elu(t) + 1 = t>0 ? t+1 : exp(t)
static __device__ __forceinline__ float phi(float x) {
    return x > 0.0f ? x + 1.0f : __expf(x);
}

// ---------------------------------------------------------------------------
// Async global->LDS staging (CDNA5 GLOBAL_LOAD_ASYNC_TO_LDS_B128, ASYNCcnt).
// Builtin prototype (from hipcc diagnostic): param0 = AS1 pointer to int
// vector-of-4 (global source), param1 = AS3 pointer (LDS dest), then two
// integer immediates (offset, cpol).
// Falls back to global->VGPR->ds_store if the builtin is unavailable.
// ---------------------------------------------------------------------------
#if defined(__has_builtin)
#if __has_builtin(__builtin_amdgcn_global_load_async_to_lds_b128)
#define HAVE_ASYNC_LDS 1
#endif
#endif

typedef int v4i_vec __attribute__((__vector_size__(16)));
typedef __attribute__((address_space(1))) v4i_vec* gptr_b128;
typedef __attribute__((address_space(3))) v4i_vec* lptr_b128;

static __device__ __forceinline__ void stage_b128(const void* g, void* l) {
#ifdef HAVE_ASYNC_LDS
    __builtin_amdgcn_global_load_async_to_lds_b128(
        (gptr_b128)(unsigned long long)(uintptr_t)g,
        (lptr_b128)(unsigned int)(uintptr_t)l,
        0, 0);
#else
    *(u32x4*)l = *(const u32x4*)g;
#endif
}

static __device__ __forceinline__ void stage_wait_all() {
#ifdef HAVE_ASYNC_LDS
#if __has_builtin(__builtin_amdgcn_s_wait_asynccnt)
    __builtin_amdgcn_s_wait_asynccnt(0);
#else
    asm volatile("s_wait_asynccnt 0x0" ::: "memory");
#endif
#endif
}

// ---------------------------------------------------------------------------
// Workspace layout (bytes)
// ---------------------------------------------------------------------------
static constexpr size_t SZ_XB   = (size_t)MROWS * FEATS * 2;        // 25165824
static constexpr size_t SZ_WT   = (size_t)4 * FEATS * FEATS * 2;    //  4718592
static constexpr size_t SZ_HEADB= (size_t)BHC * SEQ * HDIM * 2;     // 25165824
static constexpr size_t SZ_KVF  = (size_t)BHC * HDIM * HDIM * 4;    //   786432
static constexpr size_t SZ_KVB  = (size_t)BHC * HDIM * HDIM * 2;    //   393216
static constexpr size_t SZ_KSUM = (size_t)BHC * HDIM * 4;           //    12288
static constexpr size_t SZ_Z    = (size_t)BHC * SEQ * 4;            //   786432

static constexpr size_t OFF_XB     = 0;
static constexpr size_t OFF_WT     = OFF_XB + SZ_XB;
static constexpr size_t OFF_QP     = OFF_WT + SZ_WT;
static constexpr size_t OFF_KP     = OFF_QP + SZ_HEADB;
static constexpr size_t OFF_VP     = OFF_KP + SZ_HEADB;
static constexpr size_t OFF_KVF    = OFF_VP + SZ_HEADB;
static constexpr size_t OFF_KVB    = OFF_KVF + SZ_KVF;
static constexpr size_t OFF_KSUM   = OFF_KVB + SZ_KVB;
static constexpr size_t OFF_Z      = OFF_KSUM + SZ_KSUM;
static constexpr size_t OFF_NORMED = OFF_Z + SZ_Z;

// ---------------------------------------------------------------------------
// Conversion kernels
// ---------------------------------------------------------------------------
__global__ void cvt_x_kernel(const float* __restrict__ x, bf16* __restrict__ xb, int n) {
    int i = blockIdx.x * blockDim.x + threadIdx.x;
    int stride = gridDim.x * blockDim.x;
    for (; i < n; i += stride) xb[i] = f2bf(x[i]);
}

// Wt[w][n][k] = W_w[k][n]  (pre-transpose so B tiles are K-contiguous)
__global__ void cvt_w_kernel(const float* __restrict__ Wq, const float* __restrict__ Wk,
                             const float* __restrict__ Wv, const float* __restrict__ Wo,
                             bf16* __restrict__ Wt) {
    int w = blockIdx.y;
    const float* W = (w == 0) ? Wq : (w == 1) ? Wk : (w == 2) ? Wv : Wo;
    int idx = blockIdx.x * blockDim.x + threadIdx.x;   // over FEATS*FEATS
    int nn = idx / FEATS;
    int kk = idx - nn * FEATS;
    Wt[(size_t)w * FEATS * FEATS + idx] = f2bf(W[(size_t)kk * FEATS + nn]);
}

// ---------------------------------------------------------------------------
// 128x128 bf16 GEMM mainloop (K = FEATS in steps of 32), 8 waves / 256 thr.
// A row-major [*, FEATS] starting at row m0; B = Wt row-major [n][FEATS].
// Wave (id 0..7): rows 32*(id&3) .. +32, cols 64*(id>>2) .. +64 -> acc[2][4].
// Double-buffered LDS: async-stage tile kt+1 while WMMAs consume tile kt.
// ---------------------------------------------------------------------------
static __device__ __forceinline__ void gemm_tile_128x128(
    const bf16* __restrict__ Abase, const bf16* __restrict__ Bbase,
    v8f acc[2][4], bf16* As, bf16* Bs) {
    const int tid  = threadIdx.x;
    const int lane = tid & 31;
    const int waveM = (tid >> 5) & 3;
    const int waveN = (tid >> 5) >> 2;
    const int l15 = lane & 15;
    const int lhi = lane >> 4;
    constexpr int NKT = FEATS / 32;    // 24 k-tiles
    constexpr int TSZ = 128 * 32;      // bf16 elements per tile buffer

    // prologue: stage k-tile 0 into buffer 0
#pragma unroll
    for (int i = 0; i < 2; ++i) {
        int c   = tid + i * 256;       // 512 chunks of 8 bf16
        int row = c >> 2;
        int kc  = (c & 3) * 8;
        stage_b128(&Abase[(size_t)row * FEATS + kc], &As[row * 32 + kc]);
        stage_b128(&Bbase[(size_t)row * FEATS + kc], &Bs[row * 32 + kc]);
    }
    stage_wait_all();
    __syncthreads();

    for (int kt = 0; kt < NKT; ++kt) {
        const int cur = kt & 1;
        bf16* Ac = As + cur * TSZ;
        bf16* Bc = Bs + cur * TSZ;

        // stage next k-tile into the other buffer (overlaps with WMMAs below)
        if (kt + 1 < NKT) {
            bf16* An = As + (1 - cur) * TSZ;
            bf16* Bn = Bs + (1 - cur) * TSZ;
            const int k0n = (kt + 1) * 32;
#pragma unroll
            for (int i = 0; i < 2; ++i) {
                int c   = tid + i * 256;
                int row = c >> 2;
                int kc  = (c & 3) * 8;
                stage_b128(&Abase[(size_t)row * FEATS + k0n + kc], &An[row * 32 + kc]);
                stage_b128(&Bbase[(size_t)row * FEATS + k0n + kc], &Bn[row * 32 + kc]);
            }
        }

        Frag16 a[2], b[4];
#pragma unroll
        for (int i = 0; i < 2; ++i) {
            int r  = waveM * 32 + i * 16 + l15;
            int kb = lhi * 8;
            a[i].q[0] = *(const u32x4*)&Ac[r * 32 + kb];        // K kb..kb+7
            a[i].q[1] = *(const u32x4*)&Ac[r * 32 + kb + 16];   // K kb+16..kb+23
        }
#pragma unroll
        for (int j = 0; j < 4; ++j) {
            int r  = waveN * 64 + j * 16 + l15;
            int kb = lhi * 8;
            b[j].q[0] = *(const u32x4*)&Bc[r * 32 + kb];        // K kb..kb+7
            b[j].q[1] = *(const u32x4*)&Bc[r * 32 + kb + 16];   // K kb+16..kb+23
        }
#pragma unroll
        for (int i = 0; i < 2; ++i)
#pragma unroll
            for (int j = 0; j < 4; ++j)
                acc[i][j] = __builtin_amdgcn_wmma_f32_16x16x32_bf16(
                    false, a[i].v, false, b[j].v, (short)0, acc[i][j], false, false);

        stage_wait_all();
        __syncthreads();
    }
}

// ---------------------------------------------------------------------------
// QKV projection: C = x @ W_mat + bias, mat=blockIdx.z in {q,k,v}.
// phi() applied for q,k. Output scattered head-major: [b][h][n][d] bf16.
// ---------------------------------------------------------------------------
__global__ void __launch_bounds__(256) proj_qkv_kernel(
    const bf16* __restrict__ xb, const bf16* __restrict__ Wt,
    const float* __restrict__ bq, const float* __restrict__ bk,
    const float* __restrict__ bv,
    bf16* __restrict__ qp, bf16* __restrict__ kp, bf16* __restrict__ vp) {
    __shared__ __align__(16) bf16 As[2 * 128 * 32];
    __shared__ __align__(16) bf16 Bs[2 * 128 * 32];
    const int m0  = blockIdx.x * 128;
    const int n0  = blockIdx.y * 128;
    const int mat = blockIdx.z;

    v8f acc[2][4];
#pragma unroll
    for (int i = 0; i < 2; ++i)
#pragma unroll
        for (int j = 0; j < 4; ++j) acc[i][j] = 0.0f;

    gemm_tile_128x128(xb + (size_t)m0 * FEATS,
                      Wt + (size_t)mat * FEATS * FEATS + (size_t)n0 * FEATS,
                      acc, As, Bs);

    const float* bias = (mat == 0) ? bq : (mat == 1) ? bk : bv;
    bf16* dst = (mat == 0) ? qp : (mat == 1) ? kp : vp;

    const int lane = threadIdx.x & 31;
    const int waveM = (threadIdx.x >> 5) & 3;
    const int waveN = (threadIdx.x >> 5) >> 2;
    const int l15 = lane & 15;
    const int lhi = lane >> 4;

#pragma unroll
    for (int i = 0; i < 2; ++i)
#pragma unroll
        for (int j = 0; j < 4; ++j) {
            int col  = n0 + waveN * 64 + j * 16 + l15;
            int h    = col >> 6;
            int dd   = col & 63;
            float bc = bias[col];
#pragma unroll
            for (int r = 0; r < 8; ++r) {
                int m   = m0 + waveM * 32 + i * 16 + r + 8 * lhi;
                int bb  = m >> 12;       // /SEQ
                int tok = m & (SEQ - 1);
                float val = acc[i][j][r] + bc;
                if (mat < 2) val = phi(val);
                dst[(((size_t)(bb * HEADS + h)) * SEQ + tok) * HDIM + dd] = f2bf(val);
            }
        }
}

// ---------------------------------------------------------------------------
// kv[bh] += k[bh]^T @ v[bh] over a 512-token K-slice (grid.y = 8 split-K).
// 128 threads = 4 waves; wave w owns d-rows [16w,16w+16) x all 64 e-cols.
// ---------------------------------------------------------------------------
__global__ void __launch_bounds__(128) kv_kernel(
    const bf16* __restrict__ kp, const bf16* __restrict__ vp,
    float* __restrict__ kvf) {
    __shared__ __align__(16) bf16 kt[32 * 64];
    __shared__ __align__(16) bf16 vt[32 * 64];
    const int bh = blockIdx.x;
    const int nbase0 = blockIdx.y * 512;
    const int tid = threadIdx.x;
    const int lane = tid & 31;
    const int wave = tid >> 5;
    const int l15 = lane & 15;
    const int lhi = lane >> 4;
    const bf16* kbh = kp + (size_t)bh * SEQ * HDIM;
    const bf16* vbh = vp + (size_t)bh * SEQ * HDIM;

    v8f acc[4];
#pragma unroll
    for (int j = 0; j < 4; ++j) acc[j] = 0.0f;

    for (int ks = 0; ks < 16; ++ks) {
        int nb = nbase0 + ks * 32;
        __syncthreads();
#pragma unroll
        for (int i = 0; i < 2; ++i) {
            int c   = tid + i * 128;          // 256 chunks of 8 bf16
            int row = c >> 3;
            int off = (c & 7) * 8;
            stage_b128(&kbh[(size_t)(nb + row) * HDIM + off], &kt[row * 64 + off]);
            stage_b128(&vbh[(size_t)(nb + row) * HDIM + off], &vt[row * 64 + off]);
        }
        stage_wait_all();
        __syncthreads();

        // A = k^T : A row = d, A K-dim = n (transposed gather from LDS)
        Frag16 a;
        int d  = wave * 16 + l15;
        int kb = lhi * 8;
#pragma unroll
        for (int i = 0; i < 8; ++i) {
            a.e[i]     = kt[(kb + i) * 64 + d];
            a.e[i + 8] = kt[(kb + 16 + i) * 64 + d];
        }
#pragma unroll
        for (int j = 0; j < 4; ++j) {
            Frag16 b;
            int e   = j * 16 + l15;
            int kb2 = lhi * 8;
#pragma unroll
            for (int i = 0; i < 16; ++i) b.e[i] = vt[(kb2 + (i & 7) + (i >> 3) * 16) * 64 + e];
            acc[j] = __builtin_amdgcn_wmma_f32_16x16x32_bf16(
                false, a.v, false, b.v, (short)0, acc[j], false, false);
        }
    }

    float* dstb = kvf + (size_t)bh * HDIM * HDIM;
#pragma unroll
    for (int j = 0; j < 4; ++j) {
        int e = j * 16 + l15;
#pragma unroll
        for (int r = 0; r < 8; ++r) {
            int d = wave * 16 + r + 8 * lhi;
            atomicAdd(&dstb[d * HDIM + e], acc[j][r]);
        }
    }
}

// ksum[bh][d] = sum_n k[bh][n][d]
__global__ void ksum_kernel(const bf16* __restrict__ kp, float* __restrict__ ksum) {
    int bh = blockIdx.x;
    int t = threadIdx.x;                  // 256
    int d = t & 63, part = t >> 6;        // 4 partials per d
    const bf16* kb = kp + (size_t)bh * SEQ * HDIM;
    float s = 0.0f;
    for (int n = part; n < SEQ; n += 4) s += bf2f(kb[(size_t)n * HDIM + d]);
    atomicAdd(&ksum[bh * HDIM + d], s);
}

// kvb[bh][e][d] = (bf16) kvf[bh][d][e]   (transpose so num-GEMM B is K-contig)
__global__ void kvt_kernel(const float* __restrict__ kvf, bf16* __restrict__ kvb) {
    int bh = blockIdx.x;
    const float* src = kvf + (size_t)bh * HDIM * HDIM;
    bf16* dst = kvb + (size_t)bh * HDIM * HDIM;
    for (int i = threadIdx.x; i < HDIM * HDIM; i += 256) {
        int e = i >> 6, d = i & 63;
        dst[e * HDIM + d] = f2bf(src[d * HDIM + e]);
    }
}

// z[bh][n] = 1 / dot(q[bh][n][:], ksum[bh][:])
__global__ void z_kernel(const bf16* __restrict__ qp, const float* __restrict__ ksum,
                         float* __restrict__ z) {
    int rid = blockIdx.x * blockDim.x + threadIdx.x;   // < BHC*SEQ
    int bh = rid >> 12;
    int n  = rid & (SEQ - 1);
    const bf16* qrow = qp + ((size_t)bh * SEQ + n) * HDIM;
    const float* ks = ksum + bh * HDIM;
    float dot = 0.0f;
#pragma unroll
    for (int d = 0; d < HDIM; ++d) dot += bf2f(qrow[d]) * ks[d];
    z[rid] = 1.0f / dot;
}

// ---------------------------------------------------------------------------
// num = (q @ kv) * z, merged back to [m][FEATS] bf16 (normed).
// grid (BHC, SEQ/128); 8 waves; wave owns 16 rows x 64 cols (4 acc tiles).
// Fragments read directly from global (everything is L2-resident).
// ---------------------------------------------------------------------------
__global__ void __launch_bounds__(256) num_kernel(
    const bf16* __restrict__ qp, const bf16* __restrict__ kvb,
    const float* __restrict__ z, bf16* __restrict__ normed) {
    const int bh  = blockIdx.x;
    const int nb0 = blockIdx.y * 128;
    const int b = bh / HEADS, h = bh - b * HEADS;
    const int tid = threadIdx.x;
    const int lane = tid & 31;
    const int wave = tid >> 5;
    const int l15 = lane & 15;
    const int lhi = lane >> 4;
    const bf16* qbh = qp + (size_t)bh * SEQ * HDIM;
    const bf16* kvt = kvb + (size_t)bh * HDIM * HDIM;   // [e][k]

    v8f acc[4];
#pragma unroll
    for (int j = 0; j < 4; ++j) acc[j] = 0.0f;

#pragma unroll
    for (int kstep = 0; kstep < 2; ++kstep) {
        Frag16 a;
        int nrow = nb0 + wave * 16 + l15;
        int kb   = kstep * 32 + lhi * 8;
        a.q[0] = *(const u32x4*)&qbh[(size_t)nrow * HDIM + kb];
        a.q[1] = *(const u32x4*)&qbh[(size_t)nrow * HDIM + kb + 16];
#pragma unroll
        for (int j = 0; j < 4; ++j) {
            Frag16 bf;
            int e   = j * 16 + l15;
            int kb2 = kstep * 32 + lhi * 8;
            bf.q[0] = *(const u32x4*)&kvt[e * HDIM + kb2];
            bf.q[1] = *(const u32x4*)&kvt[e * HDIM + kb2 + 16];
            acc[j] = __builtin_amdgcn_wmma_f32_16x16x32_bf16(
                false, a.v, false, bf.v, (short)0, acc[j], false, false);
        }
    }

    float zr[8];
#pragma unroll
    for (int r = 0; r < 8; ++r) {
        int nrow = nb0 + wave * 16 + r + 8 * lhi;
        zr[r] = z[bh * SEQ + nrow];
    }
#pragma unroll
    for (int j = 0; j < 4; ++j) {
        int e   = j * 16 + l15;
        int col = h * HDIM + e;
#pragma unroll
        for (int r = 0; r < 8; ++r) {
            int nrow = nb0 + wave * 16 + r + 8 * lhi;
            size_t m = (size_t)b * SEQ + nrow;
            normed[m * FEATS + col] = f2bf(acc[j][r] * zr[r]);
        }
    }
}

// ---------------------------------------------------------------------------
// Output projection: out = normed @ Wo + bo (f32 out, row-major [m][FEATS]).
// ---------------------------------------------------------------------------
__global__ void __launch_bounds__(256) proj_o_kernel(
    const bf16* __restrict__ normed, const bf16* __restrict__ WtO,
    const float* __restrict__ bo, float* __restrict__ out) {
    __shared__ __align__(16) bf16 As[2 * 128 * 32];
    __shared__ __align__(16) bf16 Bs[2 * 128 * 32];
    const int m0 = blockIdx.x * 128;
    const int n0 = blockIdx.y * 128;

    v8f acc[2][4];
#pragma unroll
    for (int i = 0; i < 2; ++i)
#pragma unroll
        for (int j = 0; j < 4; ++j) acc[i][j] = 0.0f;

    gemm_tile_128x128(normed + (size_t)m0 * FEATS, WtO + (size_t)n0 * FEATS, acc, As, Bs);

    const int lane = threadIdx.x & 31;
    const int waveM = (threadIdx.x >> 5) & 3;
    const int waveN = (threadIdx.x >> 5) >> 2;
    const int l15 = lane & 15;
    const int lhi = lane >> 4;

#pragma unroll
    for (int i = 0; i < 2; ++i)
#pragma unroll
        for (int j = 0; j < 4; ++j) {
            int col  = n0 + waveN * 64 + j * 16 + l15;
            float bc = bo[col];
#pragma unroll
            for (int r = 0; r < 8; ++r) {
                int m = m0 + waveM * 32 + i * 16 + r + 8 * lhi;
                out[(size_t)m * FEATS + col] = acc[i][j][r] + bc;
            }
        }
}

// ---------------------------------------------------------------------------
// Launch
// ---------------------------------------------------------------------------
extern "C" void kernel_launch(void* const* d_in, const int* in_sizes, int n_in,
                              void* d_out, int out_size, void* d_ws, size_t ws_size,
                              hipStream_t stream) {
    const float* x  = (const float*)d_in[0];
    const float* Wq = (const float*)d_in[1];
    const float* bq = (const float*)d_in[2];
    const float* Wk = (const float*)d_in[3];
    const float* bk = (const float*)d_in[4];
    const float* Wv = (const float*)d_in[5];
    const float* bv = (const float*)d_in[6];
    const float* Wo = (const float*)d_in[7];
    const float* bo = (const float*)d_in[8];
    float* out = (float*)d_out;

    char* ws = (char*)d_ws;
    bf16*  xb     = (bf16*)(ws + OFF_XB);
    bf16*  Wt     = (bf16*)(ws + OFF_WT);
    bf16*  qp     = (bf16*)(ws + OFF_QP);
    bf16*  kp     = (bf16*)(ws + OFF_KP);
    bf16*  vp     = (bf16*)(ws + OFF_VP);
    float* kvf    = (float*)(ws + OFF_KVF);
    bf16*  kvb    = (bf16*)(ws + OFF_KVB);
    float* ksum   = (float*)(ws + OFF_KSUM);
    float* zbuf   = (float*)(ws + OFF_Z);
    bf16*  normed = (bf16*)(ws + OFF_NORMED);

    // zero the fp32 accumulators (kvf, kvb, ksum are contiguous)
    (void)hipMemsetAsync(ws + OFF_KVF, 0, SZ_KVF + SZ_KVB + SZ_KSUM, stream);

    cvt_x_kernel<<<4096, 256, 0, stream>>>(x, xb, MROWS * FEATS);
    cvt_w_kernel<<<dim3(FEATS * FEATS / 256, 4), 256, 0, stream>>>(Wq, Wk, Wv, Wo, Wt);

    proj_qkv_kernel<<<dim3(MROWS / 128, FEATS / 128, 3), 256, 0, stream>>>(
        xb, Wt, bq, bk, bv, qp, kp, vp);

    ksum_kernel<<<BHC, 256, 0, stream>>>(kp, ksum);
    kv_kernel<<<dim3(BHC, 8), 128, 0, stream>>>(kp, vp, kvf);
    kvt_kernel<<<BHC, 256, 0, stream>>>(kvf, kvb);
    z_kernel<<<BHC * SEQ / 256, 256, 0, stream>>>(qp, ksum, zbuf);

    num_kernel<<<dim3(BHC, SEQ / 128), 256, 0, stream>>>(qp, kvb, zbuf, normed);

    proj_o_kernel<<<dim3(MROWS / 128, FEATS / 128), 256, 0, stream>>>(
        normed, Wt + (size_t)3 * FEATS * FEATS, bo, out);
}
